// MultiHeadSelfAttention_81741817577579
// MI455X (gfx1250) — compile-verified, matrix operand order checked by reading, not run on hardware
//
#include <hip/hip_runtime.h>
#include <hip/hip_bf16.h>
#include <math.h>

// ---------------------------------------------------------------------------
// MI455X (gfx1250, wave32) multi-head self-attention, bf16 WMMA pipeline.
//   D = 1024, H = 16, dh = 64, tokens = 4*1024 = 4096
// ---------------------------------------------------------------------------

typedef __bf16 bf16_t;
typedef __attribute__((ext_vector_type(16))) __bf16 v16bf;
typedef __attribute__((ext_vector_type(8)))  __bf16 bf16x8;
typedef __attribute__((ext_vector_type(8)))  float  v8f;

#define TOKENS 4096
#define D      1024
#define H      16
#define DH     64
#define NS     1024
#define B3D    3072

__device__ __forceinline__ v8f wmma_bf16(v16bf a, v16bf b, v8f c) {
    return __builtin_amdgcn_wmma_f32_16x16x32_bf16(
        /*neg_a=*/false, a, /*neg_b=*/false, b,
        /*c_mod=*/(short)0, c, /*reuse_a=*/false, /*reuse_b=*/false);
}

// Load a 16x32 bf16 WMMA fragment where each lane's data comes from one row
// of a row-major matrix (used for A, and for B when B is stored transposed).
// Per-lane K map: f[2v],f[2v+1] at K = (v>>2)*16 + (lane>>4)*8 + (v&3)*2 (+kk).
// => two contiguous 16-byte loads per lane.
__device__ __forceinline__ v16bf load_frag_rows(const bf16_t* __restrict__ base,
                                                int ld, int lane, int kk) {
    const int r = lane & 15;
    const int h = lane >> 4;
    const bf16_t* p = base + (size_t)r * ld + kk + h * 8;
    bf16x8 lo = *(const bf16x8*)(p);        // K = kk + h*8 .. +7
    bf16x8 hi = *(const bf16x8*)(p + 16);   // K = kk + 16 + h*8 .. +7
    v16bf f;
#pragma unroll
    for (int i = 0; i < 8; ++i) { f[i] = lo[i]; f[8 + i] = hi[i]; }
    return f;
}

// ---------------- 16-lane butterfly reductions (VALU DPP, no LDS) ----------

#if __has_builtin(__builtin_amdgcn_update_dpp)
template <int CTRL>
__device__ __forceinline__ float dpp_movf(float x) {
    // DPP16 ROW_XMASK: lane <- lane ^ mask within each row of 16 lanes.
    return __int_as_float(__builtin_amdgcn_update_dpp(
        0, __float_as_int(x), CTRL, 0xf, 0xf, true));
}
__device__ __forceinline__ float red16_max(float x) {
    x = fmaxf(x, dpp_movf<0x161>(x));   // xor 1
    x = fmaxf(x, dpp_movf<0x162>(x));   // xor 2
    x = fmaxf(x, dpp_movf<0x164>(x));   // xor 4
    x = fmaxf(x, dpp_movf<0x168>(x));   // xor 8
    return x;
}
__device__ __forceinline__ float red16_sum(float x) {
    x += dpp_movf<0x161>(x);
    x += dpp_movf<0x162>(x);
    x += dpp_movf<0x164>(x);
    x += dpp_movf<0x168>(x);
    return x;
}
#else
__device__ __forceinline__ float red16_max(float x) {
#pragma unroll
    for (int off = 1; off < 16; off <<= 1)
        x = fmaxf(x, __shfl_xor(x, off, 32));
    return x;
}
__device__ __forceinline__ float red16_sum(float x) {
#pragma unroll
    for (int off = 1; off < 16; off <<= 1)
        x += __shfl_xor(x, off, 32);
    return x;
}
#endif

// ------------------------------- converters --------------------------------

__global__ void __launch_bounds__(256)
k_cvt_f32_bf16(const float* __restrict__ src, bf16_t* __restrict__ dst, int n) {
    int i = blockIdx.x * blockDim.x + threadIdx.x;
    if (i < n) dst[i] = (bf16_t)src[i];
}

// src[rows][cols] (f32) -> dst[cols][rows] (bf16), 32x32 LDS tiles.
__global__ void __launch_bounds__(256)
k_transpose_cvt(const float* __restrict__ src, bf16_t* __restrict__ dst,
                int rows, int cols) {
    __shared__ float tile[32][33];
    const int bx = blockIdx.x * 32;   // col base
    const int by = blockIdx.y * 32;   // row base
    const int tx = threadIdx.x;       // 0..31
    const int ty = threadIdx.y;       // 0..7
#pragma unroll
    for (int i = 0; i < 32; i += 8)
        tile[ty + i][tx] = src[(size_t)(by + ty + i) * cols + bx + tx];
    __syncthreads();
#pragma unroll
    for (int i = 0; i < 32; i += 8)
        dst[(size_t)(bx + ty + i) * rows + by + tx] = (bf16_t)tile[tx][ty + i];
}

// Extract V from qkv[4096][3072] and transpose per head:
//   Vt[b][h][d][n] = qkv[b*NS + n][2*D + h*DH + d]
__global__ void __launch_bounds__(256)
k_transpose_v(const bf16_t* __restrict__ qkv, bf16_t* __restrict__ Vt) {
    __shared__ bf16_t tile[32][34];
    const int bh = blockIdx.z;                 // b*H + h
    const int b = bh >> 4, h = bh & 15;
    const int n0 = blockIdx.y * 32;            // token within sequence
    const int d0 = blockIdx.x * 32;            // head-dim base (0 or 32)
    const int tx = threadIdx.x;                // 0..31
    const int ty = threadIdx.y;                // 0..7
    const bf16_t* src = qkv + (size_t)(b * NS + n0) * B3D + 2 * D + h * DH + d0;
#pragma unroll
    for (int i = 0; i < 32; i += 8)
        tile[ty + i][tx] = src[(size_t)(ty + i) * B3D + tx];
    __syncthreads();
    bf16_t* dst = Vt + ((size_t)bh * DH + d0) * NS + n0;
#pragma unroll
    for (int i = 0; i < 32; i += 8)
        dst[(size_t)(ty + i) * NS + tx] = tile[tx][ty + i];
}

// ------------------------------ WMMA GEMM ----------------------------------
// C[M][N] = A[M][K] @ B[K][N] + bias, with B supplied transposed (Bt[N][K]).
// Block = 256 threads (8 waves). Wave w owns a 32x64 output tile:
// rows [by*256 + 32w, +32), cols [bx*64, +64). Each B fragment feeds 2 WMMAs.
template <bool STORE_BF16>
__global__ void __launch_bounds__(256, 1)
k_gemm_bf16_wmma(const bf16_t* __restrict__ A,
                 const bf16_t* __restrict__ Bt,
                 const float* __restrict__ bias,
                 void* __restrict__ Cout,
                 int M, int N, int K) {
    const int lane = threadIdx.x & 31;
    const int wave = threadIdx.x >> 5;
    const int m0 = blockIdx.y * 256 + wave * 32;
    const int n0 = blockIdx.x * 64;

    v8f acc[2][4];
#pragma unroll
    for (int r = 0; r < 2; ++r)
#pragma unroll
        for (int t = 0; t < 4; ++t) acc[r][t] = (v8f)(0.0f);

    const bf16_t* Arow0 = A + (size_t)m0 * K;
    const bf16_t* Arow1 = Arow0 + (size_t)16 * K;
    for (int kk = 0; kk < K; kk += 32) {
        v16bf a0 = load_frag_rows(Arow0, K, lane, kk);
        v16bf a1 = load_frag_rows(Arow1, K, lane, kk);
#pragma unroll
        for (int t = 0; t < 4; ++t) {
            v16bf bfp = load_frag_rows(Bt + (size_t)(n0 + 16 * t) * K, K, lane, kk);
            acc[0][t] = wmma_bf16(a0, bfp, acc[0][t]);
            acc[1][t] = wmma_bf16(a1, bfp, acc[1][t]);
        }
    }

    const int col   = lane & 15;
    const int rbase = (lane >> 4) * 8;
#pragma unroll
    for (int t = 0; t < 4; ++t) {
        const int n = n0 + 16 * t + col;
        const float bv = bias[n];
#pragma unroll
        for (int r = 0; r < 2; ++r)
#pragma unroll
            for (int v = 0; v < 8; ++v) {
                const int m = m0 + 16 * r + rbase + v;
                const float val = acc[r][t][v] + bv;
                if (STORE_BF16) ((bf16_t*)Cout)[(size_t)m * N + n] = (bf16_t)val;
                else            ((float*)Cout)[(size_t)m * N + n] = val;
            }
    }
}

// --------------------------- flash attention -------------------------------
// qkv: bf16 [4096][3072], col = c*1024 + h*64 + d  (c = 0:q 1:k 2:v).
// Vt:  bf16 [4][16][64][1024] (per-head transposed V).
// Grid: (NS/64, H, batch). Block: 128 threads = 4 waves; each wave owns
// 16 query rows, streams 64-key tiles with online softmax.
__global__ void __launch_bounds__(128, 1)
k_attention_wmma(const bf16_t* __restrict__ qkv,
                 const bf16_t* __restrict__ Vt,
                 bf16_t* __restrict__ Obuf) {
    __shared__ bf16_t Plds[4][16][72];   // per-wave 16x64 P tile (padded)

    const int lane = threadIdx.x & 31;
    const int wave = threadIdx.x >> 5;
    const int h = blockIdx.y;
    const int b = blockIdx.z;
    const int q0 = blockIdx.x * 64 + wave * 16;          // query row in [0,1024)
    const float scale = 0.125f;                           // 1/sqrt(64)
    const int col   = lane & 15;
    const int rbase = (lane >> 4) * 8;

    const size_t tokbase = (size_t)b * NS;
    const bf16_t* Qbase  = qkv + (tokbase + q0) * B3D + h * DH;
    const bf16_t* VtHead = Vt + (size_t)(b * H + h) * DH * NS;

    // Preload Q fragments (reused across all key tiles).
    v16bf aQ0 = load_frag_rows(Qbase, B3D, lane, 0);
    v16bf aQ1 = load_frag_rows(Qbase, B3D, lane, 32);

    v8f o[4];
#pragma unroll
    for (int t = 0; t < 4; ++t) o[t] = (v8f)(0.0f);
    float mrow[8], lrow[8];
#pragma unroll
    for (int v = 0; v < 8; ++v) { mrow[v] = -INFINITY; lrow[v] = 0.0f; }

    for (int kt = 0; kt < NS; kt += 64) {
        const bf16_t* Kbase = qkv + (tokbase + kt) * B3D + D + h * DH;

        // ---- S = Q @ K^T : K rows serve directly as the "Bt" fragment ----
        v8f s[4];
#pragma unroll
        for (int t = 0; t < 4; ++t) s[t] = (v8f)(0.0f);
#pragma unroll
        for (int t = 0; t < 4; ++t) {
            v16bf bk0 = load_frag_rows(Kbase + (size_t)(16 * t) * B3D, B3D, lane, 0);
            s[t] = wmma_bf16(aQ0, bk0, s[t]);
            v16bf bk1 = load_frag_rows(Kbase + (size_t)(16 * t) * B3D, B3D, lane, 32);
            s[t] = wmma_bf16(aQ1, bk1, s[t]);
        }

        // ---- online softmax per query row v (16-lane DPP reductions).
        //      corr kept scalar; P stored to LDS as soon as it is computed,
        //      so s[t][v] and corr die early (keeps VGPR pressure down). ----
#pragma unroll
        for (int v = 0; v < 8; ++v) {
            float mx = fmaxf(fmaxf(s[0][v], s[1][v]), fmaxf(s[2][v], s[3][v]));
            mx = red16_max(mx) * scale;
            const float mnew = fmaxf(mrow[v], mx);
            const float corr = __expf(mrow[v] - mnew);
            float rs = 0.0f;
#pragma unroll
            for (int t = 0; t < 4; ++t) {
                float p = __expf(s[t][v] * scale - mnew);
                Plds[wave][rbase + v][16 * t + col] = (bf16_t)p;
                rs += p;
            }
            rs = red16_sum(rs);
            lrow[v] = lrow[v] * corr + rs;
            mrow[v] = mnew;
#pragma unroll
            for (int t = 0; t < 4; ++t) o[t][v] *= corr;
        }
        // DS ops from one wave are in-order: store->load below is safe.

        // ---- O += P @ V  (V supplied transposed -> contiguous b128 loads) --
#pragma unroll
        for (int ks = 0; ks < 2; ++ks) {
            v16bf aP = load_frag_rows(&Plds[wave][0][0], 72, lane, ks * 32);
#pragma unroll
            for (int t = 0; t < 4; ++t) {
                v16bf bv = load_frag_rows(VtHead + (size_t)(16 * t) * NS, NS,
                                          lane, kt + ks * 32);
                o[t] = wmma_bf16(aP, bv, o[t]);
            }
        }
    }

    // ---- normalize and store O as bf16 [4096][1024], col = h*64 + d ----
#pragma unroll
    for (int v = 0; v < 8; ++v) {
        const float inv = 1.0f / lrow[v];
        const size_t row = tokbase + q0 + rbase + v;
#pragma unroll
        for (int t = 0; t < 4; ++t)
            Obuf[row * D + h * DH + 16 * t + col] = (bf16_t)(o[t][v] * inv);
    }
}

// ------------------------------- launcher ----------------------------------

extern "C" void kernel_launch(void* const* d_in, const int* in_sizes, int n_in,
                              void* d_out, int out_size, void* d_ws, size_t ws_size,
                              hipStream_t stream) {
    (void)in_sizes; (void)n_in; (void)out_size; (void)ws_size;

    const float* x     = (const float*)d_in[0];   // [4,1024,1024]
    const float* W_qkv = (const float*)d_in[1];   // [1024,3072]
    const float* b_qkv = (const float*)d_in[2];   // [3072]
    const float* W_out = (const float*)d_in[3];   // [1024,1024]
    const float* b_out = (const float*)d_in[4];   // [1024]
    float* out = (float*)d_out;

    // Workspace layout (56 MiB total, all bf16):
    bf16_t* Xb    = (bf16_t*)d_ws;                       // 4096*1024
    bf16_t* WqkvT = Xb    + (size_t)TOKENS * D;          // 3072*1024
    bf16_t* WoutT = WqkvT + (size_t)B3D * D;             // 1024*1024
    bf16_t* QKVb  = WoutT + (size_t)D * D;               // 4096*3072
    bf16_t* Ob    = QKVb  + (size_t)TOKENS * B3D;        // 4096*1024
    bf16_t* Vt    = Ob    + (size_t)TOKENS * D;          // 4*16*64*1024

    // 1) convert activations / transpose-convert weights to bf16
    {
        int n = TOKENS * D;
        k_cvt_f32_bf16<<<(n + 255) / 256, 256, 0, stream>>>(x, Xb, n);
        dim3 blk(32, 8);
        k_transpose_cvt<<<dim3(B3D / 32, D / 32), blk, 0, stream>>>(W_qkv, WqkvT, D, B3D);
        k_transpose_cvt<<<dim3(D / 32, D / 32), blk, 0, stream>>>(W_out, WoutT, D, D);
    }

    // 2) QKV projection: [4096,1024] @ [1024,3072] + b_qkv -> bf16
    k_gemm_bf16_wmma<true><<<dim3(B3D / 64, TOKENS / 256), 256, 0, stream>>>(
        Xb, WqkvT, b_qkv, QKVb, TOKENS, B3D, D);

    // 2b) per-head V transpose for vectorized P@V fragments
    k_transpose_v<<<dim3(DH / 32, NS / 32, 4 * H), dim3(32, 8), 0, stream>>>(QKVb, Vt);

    // 3) attention per (batch, head, 64-query tile)
    k_attention_wmma<<<dim3(NS / 64, H, 4), 128, 0, stream>>>(QKVb, Vt, Ob);

    // 4) output projection: [4096,1024] @ [1024,1024] + b_out -> f32 d_out
    k_gemm_bf16_wmma<false><<<dim3(D / 64, TOKENS / 256), 256, 0, stream>>>(
        Ob, WoutT, b_out, out, TOKENS, D, D);
}
